// gnn_architecture_1_39049842655736
// MI455X (gfx1250) — matrix operand order repaired and checked by reading, not compiled, not run on hardware
//
#include <hip/hip_runtime.h>
#include <hip/hip_bf16.h>

// ---------------- problem constants (from reference) ----------------
#define N_NODES 100000
#define N_EDGES 600000
#define FDIM    128      // feature dim for layer1/2 in+out and layer3 in
#define BN_EPS  1e-5f
#define SLOPE   0.01f

typedef __attribute__((ext_vector_type(16))) _Float16 v16h;
typedef __attribute__((ext_vector_type(8)))  float    v8f;

// ---------------- utility: zero-fill ----------------
__global__ void fill_kernel(float* __restrict__ p, long long n) {
    long long i = (long long)blockIdx.x * blockDim.x + threadIdx.x;
    if (i < n) p[i] = 0.0f;
}

// ---------------- f32 -> f16 convert (weights, one shot per call) ----------------
__global__ void cvt_f16_kernel(const float* __restrict__ in, _Float16* __restrict__ out,
                               int n) {
    int i = blockIdx.x * blockDim.x + threadIdx.x;
    if (i < n) out[i] = (_Float16)in[i];
}

// ---------------- degree: deg[dst] += attr ----------------
__global__ void deg_kernel(const int* __restrict__ dst, const float* __restrict__ attr,
                           float* __restrict__ deg) {
    int e = blockIdx.x * blockDim.x + threadIdx.x;
    if (e < N_EDGES) atomicAdd(&deg[dst[e]], attr[e]);
}

// ---------------- dis = deg>0 ? rsqrt(max(deg,1e-12)) : 0 (in place) ----------------
__global__ void dis_kernel(float* __restrict__ deg) {
    int i = blockIdx.x * blockDim.x + threadIdx.x;
    if (i < N_NODES) {
        float d = deg[i];
        deg[i] = (d > 0.0f) ? rsqrtf(fmaxf(d, 1e-12f)) : 0.0f;
    }
}

// ---------------- norm[e] = dis[src]*attr*dis[dst] ----------------
__global__ void norm_kernel(const int* __restrict__ src, const int* __restrict__ dst,
                            const float* __restrict__ attr, const float* __restrict__ dis,
                            float* __restrict__ nrm) {
    int e = blockIdx.x * blockDim.x + threadIdx.x;
    if (e < N_EDGES) nrm[e] = dis[src[e]] * attr[e] * dis[dst[e]];
}

// ---------------- propagate: hout[dst] += norm[e] * hin[src]  (128 features) ----------
// 32 threads per edge, 4 features each -> coalesced 512B row gather per edge.
// hin/hout are 51.2 MB each: L2-resident on MI455X (192 MB L2).
__global__ void prop_kernel(const int* __restrict__ src, const int* __restrict__ dst,
                            const float* __restrict__ nrm,
                            const float* __restrict__ hin, float* __restrict__ hout) {
    int t = blockIdx.x * blockDim.x + threadIdx.x;   // E*32 = 19.2M threads, fits int
    int e = t >> 5;
    int c = (t & 31) << 2;
    if (e >= N_EDGES) return;
    float w = nrm[e];
    const float4 v = *(const float4*)(hin + (size_t)src[e] * FDIM + c);
    float* o = hout + (size_t)dst[e] * FDIM + c;
    atomicAdd(o + 0, w * v.x);
    atomicAdd(o + 1, w * v.y);
    atomicAdd(o + 2, w * v.z);
    atomicAdd(o + 3, w * v.w);
}

// ---------------- WMMA GEMM: Cout = A @ Wh (+ Cin) (+ bias) -------------------------
// A:[N,128] f32, Wh:[128, Fout] f16 (pre-converted), Fout = NCT*16.
// One wave computes a full 16 x Fout row strip: the A fragment for each K-chunk is
// loaded ONCE and reused across all NCT column tiles (8x less A traffic).
// K=128 -> 4 K-chunks; NCT WMMAs per chunk -> 4*NCT v_wmma per wave.
// Fragment layouts per CDNA5 ISA 7.12.2:
//   A 16x32 f16: lane M=l15, half=lane>>4; VGPR v pair p -> K = (v<4 ? 2v : 16+2(v-4)) + p + 8*half
//   B 32x16 f16: lane N=l15; element s -> K = (s<8 ? s : 16+(s-8)) + 8*half (strided f16 loads)
//   C/D 16x16 f32: lane N=l15; VGPR r -> M = r + 8*half
template <int NCT>
__global__ __launch_bounds__(32) void gemm_wmma(
        const float* __restrict__ A, const _Float16* __restrict__ Wh,
        const float* __restrict__ bias, const float* __restrict__ Cin,
        float* __restrict__ Cout) {
    constexpr int Fout = NCT * 16;
    const int lane = threadIdx.x;
    const int half = lane >> 4;
    const int l15  = lane & 15;
    const int row0 = blockIdx.x * 16;

    v8f c[NCT];
    if (Cin) {
        const float* cp = Cin + (size_t)(row0 + 8 * half) * Fout + l15;
        #pragma unroll
        for (int ct = 0; ct < NCT; ++ct)
            #pragma unroll
            for (int r = 0; r < 8; ++r)
                c[ct][r] = cp[(size_t)r * Fout + ct * 16];
    } else {
        #pragma unroll
        for (int ct = 0; ct < NCT; ++ct)
            #pragma unroll
            for (int r = 0; r < 8; ++r) c[ct][r] = 0.0f;
    }

    const float* arow = A + (size_t)(row0 + l15) * FDIM;   // A row for M=l15
    const int    bK   = l15 + 16 * half;                    // unused; B lane holds column N=l15

    #pragma unroll
    for (int kt = 0; kt < 4; ++kt) {
        const int k0 = kt * 32;
        v16h a;
        #pragma unroll
        for (int v = 0; v < 8; ++v) {
            int kk = k0 + ((v < 4) ? (2 * v) : (16 + 2 * (v - 4))) + 8 * half;
            float2 f2 = *(const float2*)(arow + kk);
            a[2 * v]     = (_Float16)f2.x;
            a[2 * v + 1] = (_Float16)f2.y;
        }
        const _Float16* wcol = Wh + (size_t)k0 * Fout + l15;
        #pragma unroll
        for (int ct = 0; ct < NCT; ++ct) {
            v16h b;
            #pragma unroll
            for (int s = 0; s < 16; ++s)
                b[s] = wcol[(size_t)(((s < 8) ? s : (8 + s)) + 8 * half) * Fout + ct * 16];
            // (neg_a, A, neg_b, B, c_mod, C, reuse_a, reuse_b)
            c[ct] = __builtin_amdgcn_wmma_f32_16x16x32_f16(false, a, false, b,
                                                           (short)0, c[ct],
                                                           false, false);
        }
    }

    float* op = Cout + (size_t)(row0 + 8 * half) * Fout + l15;
    #pragma unroll
    for (int ct = 0; ct < NCT; ++ct) {
        float bv = bias ? bias[ct * 16 + l15] : 0.0f;
        #pragma unroll
        for (int r = 0; r < 8; ++r)
            op[(size_t)r * Fout + ct * 16] = c[ct][r] + bv;
    }
}

// ---------------- BN stats: per-column sum and sumsq ----------------
// block = F threads (one column each), each block covers 512 rows, then 2 atomics/thread.
__global__ void bn_stats_kernel(const float* __restrict__ X, float* __restrict__ sums,
                                int F) {
    int col = threadIdx.x;
    int r0 = blockIdx.x * 512;
    int r1 = r0 + 512; if (r1 > N_NODES) r1 = N_NODES;
    float s = 0.0f, s2 = 0.0f;
    for (int r = r0; r < r1; ++r) {
        float v = X[(size_t)r * F + col];
        s += v; s2 += v * v;
    }
    atomicAdd(&sums[col], s);
    atomicAdd(&sums[F + col], s2);
}

// ---------------- BN apply + LeakyReLU ----------------
__global__ void bn_apply_kernel(const float* __restrict__ X, const float* __restrict__ sums,
                                const float* __restrict__ g, const float* __restrict__ be,
                                float* __restrict__ Y, int F, long long total) {
    long long i = (long long)blockIdx.x * blockDim.x + threadIdx.x;
    if (i >= total) return;
    int col = (int)(i & (F - 1));                 // F is 128 or 64 (power of two)
    const float invN = 1.0f / (float)N_NODES;
    float m  = sums[col] * invN;
    float vv = sums[F + col] * invN - m * m;
    float sc = g[col] * rsqrtf(vv + BN_EPS);
    float val = (X[i] - m) * sc + be[col];
    Y[i] = (val > 0.0f) ? val : SLOPE * val;
}

// =====================================================================================
extern "C" void kernel_launch(void* const* d_in, const int* in_sizes, int n_in,
                              void* d_out, int out_size, void* d_ws, size_t ws_size,
                              hipStream_t stream) {
    // Inputs in setup_inputs() order. edge_index is int32 (JAX default x64-disabled).
    const float* y    = (const float*)d_in[0];
    const int*   src  = (const int*)d_in[1];
    const int*   dst  = src + N_EDGES;
    const float* attr = (const float*)d_in[2];
    const float* W1   = (const float*)d_in[3];
    const float* b1   = (const float*)d_in[4];
    const float* g1   = (const float*)d_in[5];
    const float* be1  = (const float*)d_in[6];
    const float* W2   = (const float*)d_in[7];
    const float* b2   = (const float*)d_in[8];
    const float* g2   = (const float*)d_in[9];
    const float* be2  = (const float*)d_in[10];
    const float* W3   = (const float*)d_in[11];
    const float* g3   = (const float*)d_in[12];
    const float* be3  = (const float*)d_in[13];
    float* out = (float*)d_out;

    // workspace layout (floats); ~208 MB total
    float* ws   = (float*)d_ws;
    float* dis  = ws;                                  // N      (deg, then dis in place)
    float* nrm  = dis + N_NODES;                       // E
    float* sums = nrm + N_EDGES;                       // 256 (sum[F], sumsq[F])
    _Float16* W1h = (_Float16*)(sums + 256);           // 4*128*128 halves
    _Float16* W2h = W1h + 4 * 128 * 128;               // 4*128*128 halves
    _Float16* W3h = W2h + 4 * 128 * 128;               // 4*128*64  halves
    float* xcur = sums + 256 + (163840 / 2);           // N*128  (after 163840 halves)
    float* hA   = xcur + (size_t)N_NODES * FDIM;       // N*128
    float* hB   = hA   + (size_t)N_NODES * FDIM;       // N*128
    float* acc  = hB   + (size_t)N_NODES * FDIM;       // N*128

    auto fill = [&](float* p, long long n) {
        fill_kernel<<<(unsigned)((n + 255) / 256), 256, 0, stream>>>(p, n);
    };

    // ---- one-shot weight conversion to f16 ----
    cvt_f16_kernel<<<(4 * 128 * 128 + 255) / 256, 256, 0, stream>>>(W1, W1h, 4 * 128 * 128);
    cvt_f16_kernel<<<(4 * 128 * 128 + 255) / 256, 256, 0, stream>>>(W2, W2h, 4 * 128 * 128);
    cvt_f16_kernel<<<(4 * 128 *  64 + 255) / 256, 256, 0, stream>>>(W3, W3h, 4 * 128 * 64);

    // ---- gcn_norm: deg -> dis -> norm ----
    fill(dis, N_NODES);
    deg_kernel <<<(N_EDGES + 255) / 256, 256, 0, stream>>>(dst, attr, dis);
    dis_kernel <<<(N_NODES + 255) / 256, 256, 0, stream>>>(dis);
    norm_kernel<<<(N_EDGES + 255) / 256, 256, 0, stream>>>(src, dst, attr, dis, nrm);

    // ---- one TAGConv + BN + LeakyReLU layer ----
    auto layer = [&](const float* xin, const _Float16* Wh, const float* bias, int Fout,
                     const float* g, const float* be, float* dest) {
        const unsigned rgrid = N_NODES / 16;   // 6250 row strips, one wave each
        auto gemm = [&](const float* Ain, const _Float16* Wk, const float* bb,
                        const float* Cin, float* Cout) {
            if (Fout == 128)
                gemm_wmma<8><<<rgrid, 32, 0, stream>>>(Ain, Wk, bb, Cin, Cout);
            else
                gemm_wmma<4><<<rgrid, 32, 0, stream>>>(Ain, Wk, bb, Cin, Cout);
        };
        // hop 0: acc = xin @ W0 (+ bias folded in here)
        gemm(xin, Wh, bias, nullptr, acc);
        const float* hprev = xin;
        for (int k = 1; k <= 3; ++k) {
            float* hnext = (k & 1) ? hA : hB;
            fill(hnext, (long long)N_NODES * FDIM);
            prop_kernel<<<(N_EDGES * 32) / 256, 256, 0, stream>>>(src, dst, nrm,
                                                                  hprev, hnext);
            gemm(hnext, Wh + (size_t)k * FDIM * Fout, nullptr, acc, acc);
            hprev = hnext;
        }
        // BatchNorm (training-mode batch stats) + LeakyReLU
        fill(sums, 2 * Fout);
        bn_stats_kernel<<<(N_NODES + 511) / 512, Fout, 0, stream>>>(acc, sums, Fout);
        long long tot = (long long)N_NODES * Fout;
        bn_apply_kernel<<<(unsigned)((tot + 255) / 256), 256, 0, stream>>>(
            acc, sums, g, be, dest, Fout, tot);
    };

    layer(y,    W1h, b1,      128, g1, be1, xcur);
    layer(xcur, W2h, b2,      128, g2, be2, xcur);
    layer(xcur, W3h, nullptr,  64, g3, be3, out);
}
